// Quantize_21071109554234
// MI455X (gfx1250) — compile-verified, matrix operand order checked by reading, not run on hardware
//
#include <hip/hip_runtime.h>

// ---------------------------------------------------------------------------
// VQ-VAE quantize for MI455X (gfx1250, wave32, WMMA, async global->LDS).
//  x:     [32, 256, 64, 64] f32      embed: [1024, 256] f32
//  out:   quantize (33554432 f32) ++ embed_idx-as-f32 (131072) ++ diff (1)
//
// argmin_k ||x-e_k||^2  ==  argmax_k ( x.e_k - 0.5*||e_k||^2 )
// GEMM in bf16x3 (hi*hi + hi*lo + lo*hi, fp32 accum, 3 independent acc
// chains) on v_wmma_f32_16x16x32_bf16.  B tiles double-buffered in LDS via
// GLOBAL_LOAD_ASYNC_TO_LDS_B128 (per-lane LDS dest = free fragment swizzle).
// ---------------------------------------------------------------------------

typedef __attribute__((ext_vector_type(16))) __bf16        v16bf;
typedef __attribute__((ext_vector_type(8)))  float         v8f;
typedef __attribute__((ext_vector_type(4)))  unsigned int  v4u;
typedef __attribute__((ext_vector_type(4)))  int           v4i;

#define C_DIM 256
#define K_DIM 1024
#define HW    4096          // H*W
#define M_TOT 131072        // N*H*W
#define QSZ   33554432      // N*C*H*W

#if defined(__gfx1250__) && __has_builtin(__builtin_amdgcn_global_load_async_to_lds_b128)
#define VQ_ASYNC 1
#else
#define VQ_ASYNC 0
#endif

static __device__ __forceinline__ unsigned short f2bf_bits(float f) {
  unsigned u = __builtin_bit_cast(unsigned, f);
  u = u + 0x7FFFu + ((u >> 16) & 1u);          // round-to-nearest-even
  return (unsigned short)(u >> 16);
}
static __device__ __forceinline__ float bfbits2f(unsigned short s) {
  unsigned u = ((unsigned)s) << 16;
  return __builtin_bit_cast(float, u);
}
static __device__ __forceinline__ __bf16 bits2bf(unsigned short s) {
  return __builtin_bit_cast(__bf16, s);
}

static __device__ __forceinline__ void vq_wait_async() {
#if VQ_ASYNC
#if __has_builtin(__builtin_amdgcn_s_wait_asynccnt)
  __builtin_amdgcn_s_wait_asynccnt(0);
#else
  asm volatile("s_wait_asynccnt 0x0" ::: "memory");
#endif
#endif
}

// Stage one 16(code) x 256(feat) B tile (hi+lo bf16 planes) into `buf` in WMMA
// fragment order.  Each 16B-aligned quad of an embed row lands contiguously
// (16B aligned) in one lane's fragment slot:
//   qc&1 -> lane half, (qc>>1)&1 -> j-group, qc>>2 -> c-chunk.
// Async path: 4 x GLOBAL_LOAD_ASYNC_TO_LDS_B128 per thread (ASYNCcnt).
static __device__ __forceinline__ void stage_btile(
    const unsigned* __restrict__ ehw, const unsigned* __restrict__ elw,
    unsigned* __restrict__ buf, int kt, int tid) {
#pragma unroll
  for (int i = 0; i < 2; ++i) {
    int q   = tid + 256 * i;                 // quad index, 0..511
    int kl  = q >> 5;                        // code row in tile (B column n)
    int qc  = q & 31;                        // quad within 128-dword feat row
    int dst = (((qc >> 2) * 32) + ((qc & 1) * 16 + kl)) * 12 + ((qc >> 1) & 1) * 4;
    const unsigned* gh = ehw + (size_t)(kt * 16 + kl) * 128 + qc * 4;
    const unsigned* gl = elw + (size_t)(kt * 16 + kl) * 128 + qc * 4;
#if VQ_ASYNC
    __builtin_amdgcn_global_load_async_to_lds_b128(
        (__attribute__((address_space(1))) v4i*)gh,
        (__attribute__((address_space(3))) v4i*)(buf + dst), 0, 0);
    __builtin_amdgcn_global_load_async_to_lds_b128(
        (__attribute__((address_space(1))) v4i*)gl,
        (__attribute__((address_space(3))) v4i*)(buf + 3072 + dst), 0, 0);
#else
    v4u h = *(const v4u*)gh;
    v4u l = *(const v4u*)gl;
    *(v4u*)(buf + dst)        = h;
    *(v4u*)(buf + 3072 + dst) = l;
#endif
  }
}

// ---------------------------------------------------------------------------
// Kernel 1: split embed into bf16 hi/lo planes, 0.5*||e_k||^2, zero diff slot.
// ---------------------------------------------------------------------------
__global__ void vq_prep(const float* __restrict__ embed,
                        unsigned short* __restrict__ eh,
                        unsigned short* __restrict__ el,
                        float* __restrict__ norms,
                        float* __restrict__ diff_slot) {
  const int k = blockIdx.x;
  const int c = threadIdx.x;
  const int lane = threadIdx.x & 31;
  const int wv   = threadIdx.x >> 5;
  float e = embed[k * C_DIM + c];
  unsigned short hb = f2bf_bits(e);
  eh[k * C_DIM + c] = hb;
  el[k * C_DIM + c] = f2bf_bits(e - bfbits2f(hb));
  float v = e * e;
#pragma unroll
  for (int off = 16; off >= 1; off >>= 1) v += __shfl_xor(v, off, 32);
  __shared__ float part[8];
  if (lane == 0) part[wv] = v;
  __syncthreads();
  if (threadIdx.x == 0) {
    float s = 0.f;
#pragma unroll
    for (int i = 0; i < 8; ++i) s += part[i];
    norms[k] = 0.5f * s;
    if (k == 0) *diff_slot = 0.f;
  }
}

// ---------------------------------------------------------------------------
// Kernel 2: fused GEMM + argmax.  1024 WGs x 256 thr (8 wave32).
//   WG tile: 128 pixel rows x all K.  Wave tile: 16 rows, loops 64 k-tiles.
//   B tiles double-buffered in LDS (24 KB each); prefetch of tile 0 overlaps
//   the whole A staging/conversion phase (A staging aliases buffer 1).
// ---------------------------------------------------------------------------
__global__ void __launch_bounds__(256, 1)
vq_argmin(const float* __restrict__ x,
          const unsigned short* __restrict__ eh,
          const unsigned short* __restrict__ el,
          const float* __restrict__ norms,
          int* __restrict__ idx_out) {
  __shared__ unsigned int shb[2][6144];     // 2 x (hi 3072 | lo 3072) = 48 KB
  const int lane = threadIdx.x & 31;
  const int wv   = threadIdx.x >> 5;
  const int m0   = blockIdx.x * 128;        // first pixel row of this WG
  const int nimg = m0 >> 12;                // m0 / (H*W); 128 | 4096, no split
  const int rem0 = m0 & 4095;
  const float* xbase = x + (size_t)nimg * C_DIM * HW + rem0;
  const unsigned* ehw = (const unsigned*)eh;
  const unsigned* elw = (const unsigned*)el;

  // Prefetch B tile 0 into buffer 0 (flies while we stage/convert A).
  stage_btile(ehw, elw, shb[0], 0, threadIdx.x);

  // ---- stage x -> bf16 hi/lo A fragments, held in registers all k-loop ----
  float* af = (float*)shb[1];               // A staging aliases buffer 1
  v16bf Ahi[8], Alo[8];
  const int arow  = wv * 16 + (lane & 15);  // A: lane&15 = M row
  const int kbase = (lane >> 4) * 8;        // half-lane K offset (ISA 7.12.2)

#pragma unroll 1
  for (int cc = 0; cc < 8; ++cc) {
    __syncthreads();
#pragma unroll
    for (int i = 0; i < 16; ++i) {
      int lin = threadIdx.x + 256 * i;      // lin = c_local*128 + row
      int cl  = lin >> 7;
      int row = lin & 127;
      af[lin] = xbase[(size_t)(cc * 32 + cl) * HW + row];   // coalesced
    }
    __syncthreads();
    v16bf ah, al;
#pragma unroll
    for (int j = 0; j < 8; ++j) {
      int k0 = ((j & 4) ? 16 : 0) + (j & 3) * 2 + kbase;
      float f0 = af[k0 * 128 + arow];
      float f1 = af[(k0 + 1) * 128 + arow];
      unsigned short h0 = f2bf_bits(f0), h1 = f2bf_bits(f1);
      ah[2 * j]     = bits2bf(h0);
      ah[2 * j + 1] = bits2bf(h1);
      al[2 * j]     = bits2bf(f2bf_bits(f0 - bfbits2f(h0)));
      al[2 * j + 1] = bits2bf(f2bf_bits(f1 - bfbits2f(h1)));
    }
    Ahi[cc] = ah;
    Alo[cc] = al;
  }

  float bestv[8];
  int   bestk[8];
#pragma unroll
  for (int i = 0; i < 8; ++i) { bestv[i] = -3.4e38f; bestk[i] = 0; }
  const int ncol = lane & 15;

  // ------------------------------- k loop --------------------------------
#pragma unroll 1
  for (int kt = 0; kt < 64; ++kt) {
    vq_wait_async();                        // tile kt resident in LDS
    __syncthreads();                        // + everyone done reading shb[kt^1]
    if (kt + 1 < 64)                        // prefetch next tile, other buffer
      stage_btile(ehw, elw, shb[(kt + 1) & 1], kt + 1, threadIdx.x);

    float nv = norms[kt * 16 + ncol];       // folded in at the combine step

    // Three independent accumulator chains (3-way WMMA ILP).
    v8f a0 = {}, a1 = {}, a2 = {};
    const unsigned* base = shb[kt & 1];
#pragma unroll
    for (int cc = 0; cc < 8; ++cc) {
      const unsigned* ph = base + (cc * 32 + lane) * 12;
      const unsigned* pl = ph + 3072;
      union { v4u u[2]; v16bf b; } th, tl;
      th.u[0] = *(const v4u*)(ph);
      th.u[1] = *(const v4u*)(ph + 4);
      tl.u[0] = *(const v4u*)(pl);
      tl.u[1] = *(const v4u*)(pl + 4);
      a0 = __builtin_amdgcn_wmma_f32_16x16x32_bf16(
          false, Ahi[cc], false, th.b, (short)0, a0, false, false);
      a1 = __builtin_amdgcn_wmma_f32_16x16x32_bf16(
          false, Ahi[cc], false, tl.b, (short)0, a1, false, false);
      a2 = __builtin_amdgcn_wmma_f32_16x16x32_bf16(
          false, Alo[cc], false, th.b, (short)0, a2, false, false);
    }

    int kcol = kt * 16 + ncol;
#pragma unroll
    for (int i = 0; i < 8; ++i) {
      float v2 = (a0[i] + a1[i]) + (a2[i] - nv);
      if (v2 > bestv[i]) { bestv[i] = v2; bestk[i] = kcol; }  // strict > keeps
    }                                                          // lowest index
  }

  // ---- 16-lane butterfly argmax (offsets 1/2/4/8 stay in wave halves) ----
#pragma unroll
  for (int i = 0; i < 8; ++i) {
    float v = bestv[i];
    int   b = bestk[i];
#pragma unroll
    for (int off = 1; off < 16; off <<= 1) {
      float ov = __shfl_xor(v, off, 32);
      int   ob = __shfl_xor(b, off, 32);
      if (ov > v || (ov == v && ob < b)) { v = ov; b = ob; }   // argmin tiebreak
    }
    if ((lane & 15) == 0) {
      // C/D layout: VGPR i holds row i (lanes 0-15) and row 8+i (lanes 16-31)
      int row = wv * 16 + ((lane >> 4) << 3) + i;
      idx_out[m0 + row] = b;
    }
  }
}

// ---------------------------------------------------------------------------
// Kernel 3: gather embed[idx] back to [N,C,H,W], fused MSE, idx -> float.
// ---------------------------------------------------------------------------
__global__ void vq_gather(const float* __restrict__ x,
                          const float* __restrict__ embed,
                          const int* __restrict__ idxp,
                          float* __restrict__ outq,
                          float* __restrict__ out_idx,
                          float* __restrict__ diff_slot) {
  const int lane = threadIdx.x & 31;
  const int wv   = threadIdx.x >> 5;
  size_t t = (size_t)blockIdx.x * 256 + threadIdx.x;
  size_t o = t * 4;                           // 4 consecutive w per thread
  int n  = (int)(o >> 20);                    // o / (C*H*W)
  int c  = (int)((o >> 12) & 255);
  int hw = (int)(o & 4095);
  size_t m = ((size_t)n << 12) + (size_t)hw;

  float4 xv = *(const float4*)(x + o);
  int i0 = idxp[m], i1 = idxp[m + 1], i2 = idxp[m + 2], i3 = idxp[m + 3];
  float4 q;
  q.x = embed[(size_t)i0 * C_DIM + c];        // embed (1 MB) lives in L2
  q.y = embed[(size_t)i1 * C_DIM + c];
  q.z = embed[(size_t)i2 * C_DIM + c];
  q.w = embed[(size_t)i3 * C_DIM + c];
  *(float4*)(outq + o) = q;

  float dx = q.x - xv.x, dy = q.y - xv.y, dz = q.z - xv.z, dw = q.w - xv.w;
  float local = (dx * dx + dy * dy + dz * dz + dw * dw) * (1.0f / 33554432.0f);
#pragma unroll
  for (int off = 16; off >= 1; off >>= 1) local += __shfl_xor(local, off, 32);
  __shared__ float part[8];
  if (lane == 0) part[wv] = local;
  __syncthreads();
  if (threadIdx.x == 0) {
    float s = 0.f;
#pragma unroll
    for (int i = 0; i < 8; ++i) s += part[i];
    atomicAdd(diff_slot, s);
  }
  if (t < (size_t)M_TOT) out_idx[t] = (float)idxp[t];
}

// ---------------------------------------------------------------------------
extern "C" void kernel_launch(void* const* d_in, const int* in_sizes, int n_in,
                              void* d_out, int out_size, void* d_ws,
                              size_t ws_size, hipStream_t stream) {
  const float* x     = (const float*)d_in[0];
  const float* embed = (const float*)d_in[1];

  float* out       = (float*)d_out;
  float* out_idx   = out + (size_t)QSZ;
  float* diff_slot = out_idx + M_TOT;

  // scratch: eh (512K) | el (512K) | norms (4K) | idx (512K)  ~= 1.54 MB
  unsigned short* eh = (unsigned short*)d_ws;
  unsigned short* el = eh + (size_t)K_DIM * C_DIM;
  float* norms       = (float*)(el + (size_t)K_DIM * C_DIM);
  int*   idxp        = (int*)(norms + K_DIM);

  vq_prep<<<K_DIM, 256, 0, stream>>>(embed, eh, el, norms, diff_slot);
  vq_argmin<<<M_TOT / 128, 256, 0, stream>>>(x, eh, el, norms, idxp);
  vq_gather<<<QSZ / (256 * 4), 256, 0, stream>>>(x, embed, idxp, out, out_idx,
                                                 diff_slot);
}
